// CustomSelfAttention_6605659702063
// MI455X (gfx1250) — matrix operand order repaired and checked by reading, not compiled, not run on hardware
//
#include <hip/hip_runtime.h>

// ---------------------------------------------------------------------------
// Fused causal multi-head self-attention for gfx1250 (MI455X, wave32, WMMA).
//   B=4, S=2048, D_MODEL=1024, H=16, HD=64
// Pipeline: f32->f16 cast | WMMA proj GEMMs | flash attention (TDM-staged K,
// KV-step 64, log2-domain softmax) | WMMA out GEMM
// ---------------------------------------------------------------------------

#define BB 4
#define SS 2048
#define DD 1024
#define HH 16
#define HD 64

typedef _Float16 h8   __attribute__((ext_vector_type(8)));
typedef _Float16 h16  __attribute__((ext_vector_type(16)));
typedef float    f8   __attribute__((ext_vector_type(8)));
typedef unsigned int  u32x4 __attribute__((ext_vector_type(4)));
typedef int           i32x4 __attribute__((ext_vector_type(4)));
typedef int           i32x8 __attribute__((ext_vector_type(8)));

static __device__ inline f8 wmma_f16(h16 a, h16 b, f8 c) {
  // D = A(16x32 f16) * B(32x16 f16) + C(16x16 f32)
  return __builtin_amdgcn_wmma_f32_16x16x32_f16(
      false, a, false, b, (short)0, c, false, false);
}

// 16-half fragment as two aligned 16-byte chunks (lo at p, hi at p+hiOff).
static __device__ inline h16 ld_frag(const _Float16* p, int hiOff) {
  h8 lo = *reinterpret_cast<const h8*>(p);
  h8 hi = *reinterpret_cast<const h8*>(p + hiOff);
  return __builtin_shufflevector(lo, hi, 0,1,2,3,4,5,6,7,8,9,10,11,12,13,14,15);
}

static __device__ inline f8 zero_f8() {
  f8 z = {0.f,0.f,0.f,0.f,0.f,0.f,0.f,0.f};
  return z;
}

// ---------------------------------------------------------------------------
// TDM: DMA a 2D tile of 64-half rows (f16) into LDS with hardware padding:
// pad_interval = 32 DWORDs (one 64-half row), pad_amount = 4 DWORDs (8 halves)
// -> LDS row stride 72 halves, matching the bank-padded fragment layout.
// Descriptor per ISA cdna5 ch.8 (group0: count/lds/global/type, group1: dims).
// ---------------------------------------------------------------------------
#if __has_builtin(__builtin_amdgcn_tensor_load_to_lds)
#define HAVE_TDM 1
static __device__ inline void tdm_load_rows64(unsigned lds_addr, const void* gptr,
                                              int rows) {
  unsigned long long ga = (unsigned long long)(uintptr_t)gptr;
  u32x4 g0;
  g0[0] = 1u;                                              // count=1, user mode
  g0[1] = lds_addr;                                        // LDS byte address
  g0[2] = (unsigned)ga;                                    // global_addr[31:0]
  g0[3] = (unsigned)((ga >> 32) & 0x01FFFFFFull) | 0x80000000u; // addr[56:32]|type=2
  i32x8 g1;
  g1[0] = (1 << 16)        // data_size = 1 (2 bytes)
        | (1 << 20)        // pad_enable
        | (4 << 22)        // pad_interval: code 4 = 32 DWORDs (one row)
        | (3 << 25);       // pad_amount:  code 3 = 4 DWORDs (8 halves)
  g1[1] = (64 << 16);      // tensor_dim0[15:0] = 64 elements per row
  g1[2] = (2048 << 16);    // tensor_dim1[15:0] = 2048 rows
  g1[3] = (64 << 16);      // tile_dim0 = 64
  g1[4] = rows;            // tile_dim1
  g1[5] = 64;              // tensor_dim0_stride = 64 elements
  g1[6] = 0;
  g1[7] = 0;
  i32x4 gz = {0, 0, 0, 0};
#if defined(__clang_major__) && (__clang_major__ >= 23)
  i32x8 gz8 = {0, 0, 0, 0, 0, 0, 0, 0};
  __builtin_amdgcn_tensor_load_to_lds(g0, g1, gz, gz, gz8, 0);
#else
  __builtin_amdgcn_tensor_load_to_lds(g0, g1, gz, gz, 0);
#endif
}
#else
#define HAVE_TDM 0
#endif

// ---------------------------------------------------------------------------
// Phase 1: f32 -> f16 cast (vectorized x4)
// ---------------------------------------------------------------------------
__global__ __launch_bounds__(256)
void mha_cvt_f16(const float* __restrict__ in, _Float16* __restrict__ out,
                 unsigned long long n4) {
  unsigned long long i = (unsigned long long)blockIdx.x * blockDim.x + threadIdx.x;
  if (i < n4) {
    float4 v = reinterpret_cast<const float4*>(in)[i];
    unsigned long long o = i * 4ull;
    out[o + 0] = (_Float16)v.x;
    out[o + 1] = (_Float16)v.y;
    out[o + 2] = (_Float16)v.z;
    out[o + 3] = (_Float16)v.w;
  }
}

// ---------------------------------------------------------------------------
// Phase 2/4: WMMA GEMM  C[M,1024] = (A[M,1024] * W[1024,1024] + bias) * scale
//   Block tile 64x128, BK=32, 256 threads = 8 waves, wave tile 16x64.
//   MODE 0: f16 output scattered to [B,H,S,HD]   (QKV projections)
//   MODE 1: f32 output row-major [M,1024]        (output projection)
// LDS strides padded to 40 halves -> all fragment reads 16B-aligned ds_b128.
// ---------------------------------------------------------------------------
template <int MODE>
__global__ __launch_bounds__(256)
void mha_gemm(const _Float16* __restrict__ A,
              const _Float16* __restrict__ Bw,
              const float* __restrict__ bias, float outScale,
              _Float16* __restrict__ OutH, float* __restrict__ OutF) {
  __shared__ _Float16 sA[64 * 40];    // [m][k]
  __shared__ _Float16 sB[128 * 40];   // [n][k] (transposed tile)

  const int tid  = threadIdx.x;
  const int m0   = blockIdx.y * 64;
  const int n0   = blockIdx.x * 128;
  const int wv   = tid >> 5;
  const int lane = tid & 31;
  const int lm   = lane & 15;
  const int hb   = (lane >> 4) << 3;   // A k-group / C row offset (0 or 8)
  const int kb   = (lane >> 4) << 4;   // 0 or 16 (unused; B k offset is hb, as for A)
  const int wm   = (wv & 3) * 16;      // wave M offset within block tile
  const int wn   = (wv >> 2) * 64;     // wave N offset within block tile

  // cooperative-load indices
  const int ar = tid >> 2, ac = (tid & 3) * 8;     // A: 64 rows x 4 thr x 8 halves
  const int bk = tid >> 3, bn = (tid & 7) * 16;    // B: 32 k x 8 thr x 16 halves

  f8 acc[4];
#pragma unroll
  for (int t = 0; t < 4; ++t) acc[t] = zero_f8();

  for (int kt = 0; kt < DD; kt += 32) {
    h8 av = *reinterpret_cast<const h8*>(&A[(size_t)(m0 + ar) * DD + kt + ac]);
    *reinterpret_cast<h8*>(&sA[ar * 40 + ac]) = av;

    const _Float16* bp = &Bw[(size_t)(kt + bk) * DD + n0 + bn];
    h8 bv0 = *reinterpret_cast<const h8*>(bp);
    h8 bv1 = *reinterpret_cast<const h8*>(bp + 8);
#pragma unroll
    for (int j = 0; j < 8; ++j) {
      sB[(bn + j) * 40 + bk]     = bv0[j];
      sB[(bn + 8 + j) * 40 + bk] = bv1[j];
    }
    __syncthreads();

    h16 a = ld_frag(&sA[(wm + lm) * 40 + hb], 16);
#pragma unroll
    for (int t = 0; t < 4; ++t) {
      h16 b = ld_frag(&sB[(wn + t * 16 + lm) * 40 + hb], 16);
      acc[t] = wmma_f16(a, b, acc[t]);
    }
    __syncthreads();
  }

#pragma unroll
  for (int t = 0; t < 4; ++t) {
#pragma unroll
    for (int i = 0; i < 8; ++i) {
      const int row = m0 + wm + i + hb;            // token index (b*S+s)
      const int col = n0 + wn + t * 16 + lm;       // model dim
      const float v = (acc[t][i] + bias[col]) * outScale;
      if (MODE == 0) {
        const int b = row >> 11, s = row & (SS - 1);
        const int h = col >> 6,  hd = col & (HD - 1);
        OutH[((((size_t)b * HH + h) * SS + s) << 6) + hd] = (_Float16)v;
      } else {
        OutF[(size_t)row * DD + col] = v;
      }
    }
  }
}

// ---------------------------------------------------------------------------
// Phase 3: causal flash attention, KV step 64 (16 WMMAs per iteration).
//   Grid: (S/64, B*H). Block: 128 threads = 4 waves; wave owns 16 query rows.
//   Q was pre-scaled by (1/sqrt(1024))*log2(e) in the projection, so the
//   softmax runs natively in the exp2 domain with no per-element scaling.
//   K tile [kv][hd] stride 72 (TDM-staged with HW padding when available),
//   V tile transposed [hd][kv] stride 72, per-wave P scratch [16][64] str 72.
//   Causal mask is applied only to wave-uniformly-diagonal 16-col subtiles.
// ---------------------------------------------------------------------------
__global__ __launch_bounds__(128)
void mha_flash(const _Float16* __restrict__ Qh, const _Float16* __restrict__ Kh,
               const _Float16* __restrict__ Vh, _Float16* __restrict__ Oh) {
  __shared__ _Float16 sK[64 * 72];        // [kv][hd]
  __shared__ _Float16 sVt[64 * 72];       // [hd][kv]
  __shared__ _Float16 sP[4][16 * 72];     // per-wave P tile [qrow][kv]

  const int tid  = threadIdx.x;
  const int w    = tid >> 5;
  const int lane = tid & 31;
  const int lm   = lane & 15;
  const int hb   = (lane >> 4) << 3;      // 0 or 8
  const int kb   = (lane >> 4) << 4;      // 0 or 16

  const int    bh   = blockIdx.y;
  const size_t base = (size_t)bh * SS * HD;
  const int    qblk = blockIdx.x * 64;
  const int    qw   = qblk + w * 16;
  const int    bq   = bh >> 4, hh = bh & 15;

  // Q fragments (A-matrix layout) for the two 32-wide k-steps over HD=64.
  h16 aq0 = ld_frag(&Qh[base + (size_t)(qw + lm) * HD + 0  + hb], 16);
  h16 aq1 = ld_frag(&Qh[base + (size_t)(qw + lm) * HD + 32 + hb], 16);

  float mrow[8], lrow[8];
  f8 o[4];
#pragma unroll
  for (int i = 0; i < 8; ++i) { mrow[i] = -1.0e30f; lrow[i] = 0.f; }
#pragma unroll
  for (int t = 0; t < 4; ++t) o[t] = zero_f8();

  // cooperative staging indices: 128 threads cover 64 rows x 64 halves
  const int skv = tid >> 1;               // 0..63
  const int shc = (tid & 1) * 32;         // 0 or 32
  const int kv_end = qblk + 64;

  for (int kvb = 0; kvb < kv_end; kvb += 64) {
    // ---- stage K tile [kv][hd] (stride 72) ----
#if HAVE_TDM
    if (w == 0) {
      tdm_load_rows64((unsigned)(uintptr_t)(void*)sK,
                      &Kh[base + (size_t)kvb * HD], 64);
      __builtin_amdgcn_s_wait_tensorcnt(0);
    }
#else
    {
      const _Float16* kp = &Kh[base + (size_t)(kvb + skv) * HD + shc];
#pragma unroll
      for (int c8 = 0; c8 < 4; ++c8)
        *reinterpret_cast<h8*>(&sK[skv * 72 + shc + c8 * 8]) =
            *reinterpret_cast<const h8*>(kp + c8 * 8);
    }
#endif
    // ---- stage V tile transposed [hd][kv] (stride 72) ----
    {
      const _Float16* vp = &Vh[base + (size_t)(kvb + skv) * HD + shc];
#pragma unroll
      for (int c8 = 0; c8 < 4; ++c8) {
        h8 vv = *reinterpret_cast<const h8*>(vp + c8 * 8);
#pragma unroll
        for (int j = 0; j < 8; ++j) sVt[(shc + c8 * 8 + j) * 72 + skv] = vv[j];
      }
    }
    // prefetch next KV tile while this one is consumed
    if (kvb + 64 < kv_end) {
      __builtin_prefetch(&Kh[base + (size_t)(kvb + 64 + skv) * HD + shc], 0, 0);
      __builtin_prefetch(&Vh[base + (size_t)(kvb + 64 + skv) * HD + shc], 0, 0);
    }
    __syncthreads();

    // ---- S = Q K^T over this 16x64 kv tile (8 WMMAs) ----
    f8 sc[4];
#pragma unroll
    for (int nt = 0; nt < 4; ++nt) {
      sc[nt] = zero_f8();
      sc[nt] = wmma_f16(aq0, ld_frag(&sK[(nt * 16 + lm) * 72 + 0  + hb], 16), sc[nt]);
      sc[nt] = wmma_f16(aq1, ld_frag(&sK[(nt * 16 + lm) * 72 + 32 + hb], 16), sc[nt]);
    }

    // ---- causal mask: only diagonal subtiles (wave-uniform test) ----
#pragma unroll
    for (int nt = 0; nt < 4; ++nt) {
      if (kvb + nt * 16 + 15 > qw) {                 // uniform per wave
        const int kg = kvb + nt * 16 + lm;
#pragma unroll
        for (int i = 0; i < 8; ++i)
          if (kg > qw + i + hb) sc[nt][i] = -1.0e30f;
      }
    }

    // ---- online softmax in exp2 domain (rows split across 16-lane halves) ----
#pragma unroll
    for (int i = 0; i < 8; ++i) {
      float mx = fmaxf(fmaxf(sc[0][i], sc[1][i]), fmaxf(sc[2][i], sc[3][i]));
      mx = fmaxf(mx, __shfl_xor(mx, 1));
      mx = fmaxf(mx, __shfl_xor(mx, 2));
      mx = fmaxf(mx, __shfl_xor(mx, 4));
      mx = fmaxf(mx, __shfl_xor(mx, 8));
      const float mnew  = fmaxf(mrow[i], mx);
      const float alpha = exp2f(mrow[i] - mnew);
      const float p0 = exp2f(sc[0][i] - mnew);
      const float p1 = exp2f(sc[1][i] - mnew);
      const float p2 = exp2f(sc[2][i] - mnew);
      const float p3 = exp2f(sc[3][i] - mnew);

      float rs = (p0 + p1) + (p2 + p3);
      rs += __shfl_xor(rs, 1);
      rs += __shfl_xor(rs, 2);
      rs += __shfl_xor(rs, 4);
      rs += __shfl_xor(rs, 8);

      lrow[i] = lrow[i] * alpha + rs;
      mrow[i] = mnew;
      o[0][i] *= alpha; o[1][i] *= alpha; o[2][i] *= alpha; o[3][i] *= alpha;

      const int pr = (i + hb) * 72;
      sP[w][pr + lm]      = (_Float16)p0;
      sP[w][pr + 16 + lm] = (_Float16)p1;
      sP[w][pr + 32 + lm] = (_Float16)p2;
      sP[w][pr + 48 + lm] = (_Float16)p3;
    }

    // wave-private LDS round-trip: P stores must land before fragment reads
    asm volatile("s_wait_dscnt 0" ::: "memory");

    // ---- O += P V  (A = P 16x64 in two k-steps, B = V^T tiles; 8 WMMAs) ----
    h16 ap0 = ld_frag(&sP[w][lm * 72 + hb], 16);
    h16 ap1 = ld_frag(&sP[w][lm * 72 + 32 + hb], 16);
#pragma unroll
    for (int t = 0; t < 4; ++t) {
      o[t] = wmma_f16(ap0, ld_frag(&sVt[(t * 16 + lm) * 72 + 0  + hb], 16), o[t]);
      o[t] = wmma_f16(ap1, ld_frag(&sVt[(t * 16 + lm) * 72 + 32 + hb], 16), o[t]);
    }
    __syncthreads();   // protect sK/sVt before next iteration's staging
  }

  // ---- epilogue: normalize, scatter to [B*S, D] f16 (head-concatenated) ----
#pragma unroll
  for (int t = 0; t < 4; ++t) {
#pragma unroll
    for (int i = 0; i < 8; ++i) {
      const int row = bq * SS + qw + i + hb;
      const int col = hh * HD + t * 16 + lm;
      Oh[(size_t)row * DD + col] = (_Float16)(o[t][i] / lrow[i]);
    }
  }
}

// ---------------------------------------------------------------------------
// Host launcher
// ---------------------------------------------------------------------------
extern "C" void kernel_launch(void* const* d_in, const int* /*in_sizes*/, int /*n_in*/,
                              void* d_out, int /*out_size*/, void* d_ws, size_t /*ws_size*/,
                              hipStream_t stream) {
  const float* q  = (const float*)d_in[0];
  const float* k  = (const float*)d_in[1];
  const float* v  = (const float*)d_in[2];
  const float* Wq = (const float*)d_in[3];
  const float* bq = (const float*)d_in[4];
  const float* Wk = (const float*)d_in[5];
  const float* bk = (const float*)d_in[6];
  const float* Wv = (const float*)d_in[7];
  const float* bv = (const float*)d_in[8];
  const float* Wo = (const float*)d_in[9];
  const float* bo = (const float*)d_in[10];

  const size_t NX = (size_t)BB * SS * DD;   // 8,388,608 activation elements
  const size_t NW = (size_t)DD * DD;        // 1,048,576 weight elements

  _Float16* ws  = (_Float16*)d_ws;
  _Float16* Xq  = ws;
  _Float16* Xk  = Xq  + NX;
  _Float16* Xv  = Xk  + NX;
  _Float16* Wqh = Xv  + NX;
  _Float16* Wkh = Wqh + NW;
  _Float16* Wvh = Wkh + NW;
  _Float16* Woh = Wvh + NW;
  _Float16* Qh  = Woh + NW;      // [B,H,S,HD]
  _Float16* Kh  = Qh  + NX;
  _Float16* Vh  = Kh  + NX;
  _Float16* AO  = Vh  + NX;      // attention output, [B*S, D]

  // Phase 1: casts
  {
    const unsigned long long x4 = NX / 4, w4 = NW / 4;
    const int blk = 256;
    mha_cvt_f16<<<(unsigned)((x4 + blk - 1) / blk), blk, 0, stream>>>(q,  Xq,  x4);
    mha_cvt_f16<<<(unsigned)((x4 + blk - 1) / blk), blk, 0, stream>>>(k,  Xk,  x4);
    mha_cvt_f16<<<(unsigned)((x4 + blk - 1) / blk), blk, 0, stream>>>(v,  Xv,  x4);
    mha_cvt_f16<<<(unsigned)((w4 + blk - 1) / blk), blk, 0, stream>>>(Wq, Wqh, w4);
    mha_cvt_f16<<<(unsigned)((w4 + blk - 1) / blk), blk, 0, stream>>>(Wk, Wkh, w4);
    mha_cvt_f16<<<(unsigned)((w4 + blk - 1) / blk), blk, 0, stream>>>(Wv, Wvh, w4);
    mha_cvt_f16<<<(unsigned)((w4 + blk - 1) / blk), blk, 0, stream>>>(Wo, Woh, w4);
  }

  // Phase 2: Q/K/V projections; Q is pre-scaled by (1/32)*log2(e) so the
  // attention softmax runs natively in the exp2 domain.
  {
    dim3 grid(DD / 128, (BB * SS) / 64);   // (8, 128)
    const float qscale = 0.04508422f;      // (1/sqrt(1024)) * log2(e)
    mha_gemm<0><<<grid, 256, 0, stream>>>(Xq, Wqh, bq, qscale, Qh, nullptr);
    mha_gemm<0><<<grid, 256, 0, stream>>>(Xk, Wkh, bk, 1.0f,   Kh, nullptr);
    mha_gemm<0><<<grid, 256, 0, stream>>>(Xv, Wvh, bv, 1.0f,   Vh, nullptr);
  }

  // Phase 3: causal flash attention
  {
    dim3 grid(SS / 64, BB * HH);           // (32, 64)
    mha_flash<<<grid, 128, 0, stream>>>(Qh, Kh, Vh, AO);
  }

  // Phase 4: output projection -> f32 d_out
  {
    dim3 grid(DD / 128, (BB * SS) / 64);
    mha_gemm<1><<<grid, 256, 0, stream>>>(AO, Woh, bo, 1.0f, nullptr, (float*)d_out);
  }
}
